// Highway_Concat_BiLSTM_61254823575735
// MI455X (gfx1250) — matrix operand order repaired and checked by reading, not compiled, not run on hardware
//
#include <hip/hip_runtime.h>
#include <hip/hip_bf16.h>

// ---------------------------------------------------------------------------
// Highway_Concat_BiLSTM for MI455X (gfx1250, wave32, WMMA)
// T=512 B=64 D=800 H=400 L=3
// ---------------------------------------------------------------------------

typedef __attribute__((ext_vector_type(16))) __bf16 v16bf;
typedef __attribute__((ext_vector_type(8)))  float  v8f;

#define T_   512
#define B_   64
#define D_   800
#define H_   400
#define NL   3
#define G3H  1200          // 3*H
#define DK   800           // input-GEMM K (== D)
#define KP   416           // recurrent K padded to 13*32
#define WST  1216          // bf16 weight row stride (800 + 416, pad zeroed)
#define ROWS (T_ * B_)     // 32768

// ---- WMMA fragment loaders (layouts per CDNA5 ISA 7.12.2, wave32) ----------

// A: 16x32 bf16, lane(<16)=row M, elem j -> k = k0 + (j>=8?16:0) + half*8 + j%8
__device__ __forceinline__ v16bf load_a_frag(const __bf16* base, int lda,
                                             int row0, int k0, int lane) {
  const int half = (lane >> 4) & 1;
  const int r    = lane & 15;
  const __bf16* p = base + (size_t)(row0 + r) * lda + k0 + half * 8;
  union { uint4 u; __bf16 h[8]; } lo, hi;
  lo.u = *(const uint4*)p;
  hi.u = *(const uint4*)(p + 16);
  v16bf a;
#pragma unroll
  for (int j = 0; j < 8; ++j) { a[j] = lo.h[j]; a[j + 8] = hi.h[j]; }
  return a;
}

// B: 32x16 bf16, lane%16 = column N, elem j -> k = k0 + (j>=8?16:0) + half*8 + j%8.
// B[k][n] == W[n][k] (row-major weights), so each lane reads two 16B chunks.
__device__ __forceinline__ v16bf load_b_frag(const __bf16* base, int ldb,
                                             int n0, int k0, int lane) {
  const int half = (lane >> 4) & 1;
  const int n    = lane & 15;
  const __bf16* p = base + (size_t)(n0 + n) * ldb + k0 + half * 8;
  union { uint4 u; __bf16 h[8]; } lo, hi;
  lo.u = *(const uint4*)p;
  hi.u = *(const uint4*)(p + 16);
  v16bf b;
#pragma unroll
  for (int j = 0; j < 8; ++j) { b[j] = lo.h[j]; b[j + 8] = hi.h[j]; }
  return b;
}

__device__ __forceinline__ v8f wmma_bf16(v16bf a, v16bf b, v8f c) {
  return __builtin_amdgcn_wmma_f32_16x16x32_bf16(false, a, false, b,
                                                 (short)0, c, false, false);
}

__device__ __forceinline__ float sigmoidf_(float x) {
  return 1.0f / (1.0f + __expf(-x));
}

// ---- conversion kernels ----------------------------------------------------

__global__ void cvt_f32_bf16(const float* __restrict__ src,
                             __bf16* __restrict__ dst, int n) {
  for (int i = blockIdx.x * blockDim.x + threadIdx.x; i < n;
       i += gridDim.x * blockDim.x)
    dst[i] = (__bf16)src[i];
}

// fW/bW [L,3H,1200] -> bf16 padded [L*2, 3H, WST] (cols >=1200 zero)
__global__ void cvt_weights(const float* __restrict__ fW,
                            const float* __restrict__ bW,
                            __bf16* __restrict__ wbf) {
  const int total = NL * 2 * G3H * WST;
  for (int i = blockIdx.x * blockDim.x + threadIdx.x; i < total;
       i += gridDim.x * blockDim.x) {
    int k  = i % WST;
    int n  = (i / WST) % G3H;
    int ld = i / (WST * G3H);       // layer*2 + dir
    int layer = ld >> 1, dir = ld & 1;
    const float* src = dir ? bW : fW;
    float v = (k < (D_ + H_)) ? src[((size_t)layer * G3H + n) * (D_ + H_) + k]
                              : 0.0f;
    wbf[i] = (__bf16)v;
  }
}

// ---- pre-gate GEMM:  G[dir][row][n] = Xbf @ Wx.T + bias --------------------
// 64x48 register-blocked macro-tile per wave (4M x 3N = 12 accumulators).
// Per K-step: 4 A-frags + 3 B-frags feed 12 WMMAs (3.4x less L2 traffic than
// one-tile-per-wave).

__global__ void __launch_bounds__(256)
pregate_gemm(const __bf16* __restrict__ xbf, const __bf16* __restrict__ wbf,
             const float* __restrict__ fb, const float* __restrict__ bb,
             float* __restrict__ G, int layer) {
  const int lane = threadIdx.x & 31;
  const int gid  = blockIdx.x * 8 + (threadIdx.x >> 5);
  const int NBL = 25;                      // 75 N-tiles / 3
  const int MBL = 512;                     // 2048 M-tiles / 4
  const int nb  = gid % NBL;
  const int mb  = (gid / NBL) % MBL;
  const int dir = gid / (NBL * MBL);
  if (dir >= 2) return;

  const __bf16* W = wbf + (size_t)((layer * 2 + dir) * G3H) * WST;
  const float* bias = (dir == 0 ? fb : bb) + layer * G3H;

  const int row0 = mb * 64, n0 = nb * 48;
  v8f acc[4][3] = {};
  for (int kt = 0; kt < DK / 32; ++kt) {
    v16bf a[4];
#pragma unroll
    for (int mm = 0; mm < 4; ++mm)
      a[mm] = load_a_frag(xbf, DK, row0 + mm * 16, kt * 32, lane);
#pragma unroll
    for (int nn = 0; nn < 3; ++nn) {
      v16bf b = load_b_frag(W, WST, n0 + nn * 16, kt * 32, lane);
#pragma unroll
      for (int mm = 0; mm < 4; ++mm)
        acc[mm][nn] = wmma_bf16(a[mm], b, acc[mm][nn]);
    }
  }
  const int ln = lane & 15, half = (lane >> 4) & 1;
  float* g = G + (size_t)dir * ROWS * G3H;
#pragma unroll
  for (int nn = 0; nn < 3; ++nn) {
    const int n  = n0 + nn * 16 + ln;
    const float bv = bias[n];
#pragma unroll
    for (int mm = 0; mm < 4; ++mm) {
      const int b0 = row0 + mm * 16 + half * 8;
#pragma unroll
      for (int i = 0; i < 8; ++i)
        g[(size_t)(b0 + i) * G3H + n] = acc[mm][nn][i] + bv;
    }
  }
}

// ---- serial LSTM scan: one workgroup per direction -------------------------
// 25 waves; wave w owns hidden columns [16w,16w+16) for all 3 gates and all
// 4 batch tiles. h lives in LDS (bf16, padded K=416), c lives in registers.
// K loop outermost so each gate's B-fragment is loaded once and reused across
// all 4 batch tiles (4x cut in the serial-phase L2 traffic).

__global__ void __launch_bounds__(800)
lstm_scan(const float* __restrict__ G, const __bf16* __restrict__ wbf,
          const float* __restrict__ f_init, const float* __restrict__ b_init,
          const float* __restrict__ masks, float* __restrict__ scanOut,
          float* __restrict__ hn, float* __restrict__ cn, int layer) {
  __shared__ __align__(16) __bf16 hbf[B_][KP];   // 53 KB

  const int dir  = blockIdx.x;
  const int tid  = threadIdx.x;
  const int lane = tid & 31;
  const int nh   = tid >> 5;                     // 0..24
  const int half = lane >> 4;
  const int ln   = lane & 15;
  const int col  = nh * 16 + ln;

  const __bf16* W  = wbf + (size_t)((layer * 2 + dir) * G3H) * WST;
  const float* init = (dir == 0 ? f_init : b_init) + layer * 2 * H_;
  const float* Gd   = G + (size_t)dir * ROWS * G3H;

  for (int idx = tid; idx < B_ * KP; idx += blockDim.x) {
    int k = idx % KP;
    hbf[idx / KP][k] = (k < H_) ? (__bf16)init[k] : (__bf16)0.0f;
  }

  float c_reg[4][8], h_reg[4][8];
#pragma unroll
  for (int m = 0; m < 4; ++m)
#pragma unroll
    for (int i = 0; i < 8; ++i) c_reg[m][i] = init[H_ + col];
  __syncthreads();

  for (int s = 0; s < T_; ++s) {
    const int t = dir ? (T_ - 1 - s) : s;
    const float* gt = Gd + (size_t)t * B_ * G3H;

    // prefetch next step's pre-gate rows (streamed once) behind this step's
    // WMMA chain -> global_prefetch_b8
    if (s + 1 < T_) {
      const float* gnext = Gd + (size_t)(dir ? t - 1 : t + 1) * B_ * G3H;
      __builtin_prefetch(gnext + tid * 96, 0, 1);    // 800*384B = full slab
    }

    v8f accI[4], accJ[4], accO[4];
#pragma unroll
    for (int m = 0; m < 4; ++m) {                // pre-gates from global
      const int b0 = m * 16 + half * 8;
#pragma unroll
      for (int i = 0; i < 8; ++i) {
        const float* gr = gt + (size_t)(b0 + i) * G3H + col;
        accI[m][i] = gr[0]; accJ[m][i] = gr[H_]; accO[m][i] = gr[2 * H_];
      }
    }
#pragma unroll
    for (int kt = 0; kt < KP / 32; ++kt) {       // 13 K-steps
      v16bf bi = load_b_frag(W, WST, nh * 16,          DK + kt * 32, lane);
      v16bf bj = load_b_frag(W, WST, H_ + nh * 16,     DK + kt * 32, lane);
      v16bf bo = load_b_frag(W, WST, 2 * H_ + nh * 16, DK + kt * 32, lane);
#pragma unroll
      for (int m = 0; m < 4; ++m) {              // B reused 4x, 12 WMMA/K-step
        v16bf a = load_a_frag(&hbf[0][0], KP, m * 16, kt * 32, lane);
        accI[m] = wmma_bf16(a, bi, accI[m]);
        accJ[m] = wmma_bf16(a, bj, accJ[m]);
        accO[m] = wmma_bf16(a, bo, accO[m]);
      }
    }
#pragma unroll
    for (int m = 0; m < 4; ++m) {                // cell math in registers
      const int b0 = m * 16 + half * 8;
#pragma unroll
      for (int i = 0; i < 8; ++i) {
        const float mt = masks[t * B_ + b0 + i];
        const float ig = sigmoidf_(accI[m][i]);
        const float cc = mt * ((1.0f - ig) * c_reg[m][i]
                               + ig * tanhf(accJ[m][i]));
        const float hh = mt * (tanhf(cc) * sigmoidf_(accO[m][i]));
        c_reg[m][i] = cc;
        h_reg[m][i] = hh;
        scanOut[(size_t)(t * B_ + b0 + i) * (2 * H_) + dir * H_ + col] = hh;
      }
    }
    __syncthreads();                             // all hbf reads done
#pragma unroll
    for (int m = 0; m < 4; ++m) {
      const int b0 = m * 16 + half * 8;
#pragma unroll
      for (int i = 0; i < 8; ++i)
        hbf[b0 + i][col] = (__bf16)h_reg[m][i];
    }
    __syncthreads();                             // new h visible
  }

  const int slot = 2 * layer + dir;
#pragma unroll
  for (int m = 0; m < 4; ++m) {
    const int b0 = m * 16 + half * 8;
#pragma unroll
    for (int i = 0; i < 8; ++i) {
      hn[(size_t)slot * B_ * H_ + (b0 + i) * H_ + col] = h_reg[m][i];
      cn[(size_t)slot * B_ * H_ + (b0 + i) * H_ + col] = c_reg[m][i];
    }
  }
}

// ---- highway: g = sigmoid(out @ pW.T + pb), 4M x 2N register blocked -------

__global__ void __launch_bounds__(256)
hw_gate(const __bf16* __restrict__ obf, const __bf16* __restrict__ pwbf,
        const float* __restrict__ pb, float* __restrict__ gbuf, int layer) {
  const int lane = threadIdx.x & 31;
  const int gid  = blockIdx.x * 8 + (threadIdx.x >> 5);
  const int NBL = 25;                      // 50 N-tiles / 2
  const int nb  = gid % NBL;
  const int mb  = gid / NBL;               // < 512
  if (mb >= ROWS / 64) return;

  const __bf16* PW = pwbf + (size_t)(layer - 1) * (2 * H_) * (2 * H_);
  const int row0 = mb * 64, n0 = nb * 32;
  v8f acc[4][2] = {};
  for (int kt = 0; kt < (2 * H_) / 32; ++kt) {   // 25
    v16bf a[4];
#pragma unroll
    for (int mm = 0; mm < 4; ++mm)
      a[mm] = load_a_frag(obf, 2 * H_, row0 + mm * 16, kt * 32, lane);
#pragma unroll
    for (int nn = 0; nn < 2; ++nn) {
      v16bf b = load_b_frag(PW, 2 * H_, n0 + nn * 16, kt * 32, lane);
#pragma unroll
      for (int mm = 0; mm < 4; ++mm)
        acc[mm][nn] = wmma_bf16(a[mm], b, acc[mm][nn]);
    }
  }
  const int ln = lane & 15, half = (lane >> 4) & 1;
#pragma unroll
  for (int nn = 0; nn < 2; ++nn) {
    const int n  = n0 + nn * 16 + ln;
    const float bv = pb[(layer - 1) * 2 * H_ + n];
#pragma unroll
    for (int mm = 0; mm < 4; ++mm) {
      const int b0 = row0 + mm * 16 + half * 8;
#pragma unroll
      for (int i = 0; i < 8; ++i)
        gbuf[(size_t)(b0 + i) * (2 * H_) + n] = sigmoidf_(acc[mm][nn][i] + bv);
    }
  }
}

__global__ void hw_mix(const float* __restrict__ g, const float* __restrict__ so,
                       const float* __restrict__ inp, float* __restrict__ out,
                       int n) {
  for (int i = blockIdx.x * blockDim.x + threadIdx.x; i < n;
       i += gridDim.x * blockDim.x) {
    const float gg = g[i];
    out[i] = gg * so[i] + (1.0f - gg) * inp[i];
  }
}

// ---------------------------------------------------------------------------

extern "C" void kernel_launch(void* const* d_in, const int* in_sizes, int n_in,
                              void* d_out, int out_size, void* d_ws,
                              size_t ws_size, hipStream_t stream) {
  (void)in_sizes; (void)n_in; (void)out_size; (void)ws_size;
  const float* x      = (const float*)d_in[0];
  const float* masks  = (const float*)d_in[1];
  const float* fW     = (const float*)d_in[2];
  const float* fb     = (const float*)d_in[3];
  const float* bW     = (const float*)d_in[4];
  const float* bb     = (const float*)d_in[5];
  const float* f_init = (const float*)d_in[6];
  const float* b_init = (const float*)d_in[7];
  const float* pW     = (const float*)d_in[8];
  const float* pb     = (const float*)d_in[9];

  float* out = (float*)d_out;
  float* hn  = out + (size_t)ROWS * 2 * H_;
  float* cn  = hn + 2 * NL * B_ * H_;

  char* ws = (char*)d_ws;
  auto carve = [&](size_t bytes) -> char* {
    char* p = ws; ws += (bytes + 255) & ~(size_t)255; return p;
  };
  __bf16* wbf  = (__bf16*)carve(sizeof(__bf16) * NL * 2 * G3H * WST);
  __bf16* pwbf = (__bf16*)carve(sizeof(__bf16) * 2 * (2 * H_) * (2 * H_));
  __bf16* xbf  = (__bf16*)carve(sizeof(__bf16) * (size_t)ROWS * DK);
  float*  G    = (float*) carve(sizeof(float) * 2ull * ROWS * G3H);
  float*  actA = (float*) carve(sizeof(float) * (size_t)ROWS * 2 * H_);
  float*  actB = (float*) carve(sizeof(float) * (size_t)ROWS * 2 * H_);

  cvt_weights<<<2048, 256, 0, stream>>>(fW, bW, wbf);
  cvt_f32_bf16<<<1280, 256, 0, stream>>>(pW, pwbf, 2 * (2 * H_) * (2 * H_));

  const float* inP = x;
  for (int layer = 0; layer < NL; ++layer) {
    cvt_f32_bf16<<<8192, 256, 0, stream>>>(inP, xbf, ROWS * DK);
    pregate_gemm<<<3200, 256, 0, stream>>>(xbf, wbf, fb, bb, G, layer);

    float* so = (layer == 1) ? actB : actA;      // layer 0,2 -> actA; 1 -> actB
    lstm_scan<<<2, 800, 0, stream>>>(G, wbf, f_init, b_init, masks, so, hn, cn,
                                     layer);
    if (layer > 0) {
      cvt_f32_bf16<<<8192, 256, 0, stream>>>(so, xbf, ROWS * 2 * H_);
      hw_gate<<<1600, 256, 0, stream>>>(xbf, pwbf, pb, G, layer);
      float* dst = (layer == NL - 1) ? out : so; // pass-2 is pure elementwise
      hw_mix<<<8192, 256, 0, stream>>>(G, so, inP, dst, ROWS * 2 * H_);
      inP = dst;
    } else {
      inP = so;
    }
  }
}
